// IBCNN_70282844831999
// MI455X (gfx1250) — matrix operand order repaired and checked by reading, not compiled, not run on hardware
//
#include <hip/hip_runtime.h>
#include <hip/hip_bf16.h>

// ---------------------------------------------------------------------------
// MI455X (gfx1250) implementation.
// Dominant cost: 10 Newton-Schulz iterations of batched 512x512x512 GEMMs
// (~515 GFLOP) -> bf16 WMMA (v_wmma_f32_16x16x32_bf16), f32 accumulate.
// All NS operands are polynomials in the symmetric Gram matrix => symmetric
// and commuting, so WMMA B-operand columns are just rows (no transposes).
// Data movement: GLOBAL_LOAD_ASYNC_TO_LDS_B128 (ASYNCcnt) into double-
// buffered LDS tiles, overlapping HBM/L2 traffic with WMMA issue.
// Steady-state k-loop is branch-free: wait(4) -> compute -> issue(next+1).
// ---------------------------------------------------------------------------

typedef __attribute__((ext_vector_type(16))) __bf16 v16bf;
typedef __attribute__((ext_vector_type(8)))  __bf16 v8bf;
typedef __attribute__((ext_vector_type(8)))  float  v8f;
typedef __attribute__((ext_vector_type(4)))  int    v4i;

#define DIM      512
#define SPATIAL  784
#define KPAD     800          // 784 padded to multiple of 32
#define NBATCH   64
#define NOUT     200
#define BM       128
#define BN       128
#define BK       32
#define LDSL     (BK + 8)     // +8 bf16 pad to spread LDS banks

// ---- CDNA5 async global->LDS copy + ASYNCcnt wait -------------------------
#if __has_builtin(__builtin_amdgcn_global_load_async_to_lds_b128)
#define ASYNC_B128(g, l)                                                      \
    __builtin_amdgcn_global_load_async_to_lds_b128(                           \
        (__attribute__((address_space(1))) v4i*)(uintptr_t)(g),               \
        (__attribute__((address_space(3))) v4i*)(uint32_t)(uintptr_t)(l),     \
        0, 0)
#else
#define ASYNC_B128(g, l)                                                      \
    asm volatile("global_load_async_to_lds_b128 %0, %1, off"                  \
                 :: "v"((uint32_t)(uintptr_t)(l)),                            \
                    "v"((uint64_t)(uintptr_t)(g))                             \
                 : "memory")
#endif

#if __has_builtin(__builtin_amdgcn_s_wait_asynccnt)
#define WAIT_ASYNC(n) __builtin_amdgcn_s_wait_asynccnt(n)
#else
#define WAIT_ASYNC(n) asm volatile("s_wait_asynccnt %0" :: "n"(n) : "memory")
#endif

union Frag { v16bf v; v8bf h[2]; };

// ---------------------------------------------------------------------------
// Generic batched bf16 GEMM:  C = alpha*(A x B) + beta*I   (M=N=512)
// B is passed as a matrix whose ROWS are the columns of the mathematical B
// (identical to B itself for all our symmetric operands; equals x for Gram).
// Optional: accumulate sum((alpha*AB)^2) per batch into normsq (Gram norm).
// Batches z < split read A0/C0, z >= split read A1/C1 (fused Ynew/Znew pass).
// ---------------------------------------------------------------------------
__global__ __launch_bounds__(256)
void gemm_bf16_wmma(const __bf16* __restrict__ A0, const __bf16* __restrict__ A1,
                    const __bf16* __restrict__ Bm,
                    __bf16* __restrict__ C0, __bf16* __restrict__ C1,
                    int split, int K, int lda, int ldb,
                    float alpha, float beta, float* __restrict__ normsq)
{
    __shared__ __align__(16) __bf16 As[2][BM][LDSL];   // 2 x 10 KB
    __shared__ __align__(16) __bf16 Bs[2][BN][LDSL];   // 2 x 10 KB
    __shared__ float red[256];

    const int tid  = threadIdx.x;
    const int lane = tid & 31;          // wave32
    const int wave = tid >> 5;          // 8 waves
    const int wrow = wave >> 2;         // 2 wave-rows  (64 M each)
    const int wcol = wave & 3;          // 4 wave-cols  (32 N each)
    const int half = lane >> 4;         // lane group 0/1
    const int lr   = lane & 15;

    const int  z      = blockIdx.z;
    const bool second = (z >= split);
    const int  b      = second ? (z - split) : z;

    const __bf16* A  = (second ? A1 : A0) + (size_t)b * (size_t)DIM * (size_t)lda;
    const __bf16* Bp = Bm                 + (size_t)b * (size_t)DIM * (size_t)ldb;
    __bf16*       C  = (second ? C1 : C0) + (size_t)b * (size_t)DIM * (size_t)DIM;

    const int m0 = blockIdx.y * BM;
    const int n0 = blockIdx.x * BN;

    v8f acc[4][2] = {};                 // 64 f32 accum VGPRs / lane

    const int r = tid >> 1;             // 0..127 cooperative-load row
    const int c = (tid & 1) * 16;       // 0/16   cooperative-load col

    const __bf16* gArow = A  + (size_t)(m0 + r) * lda + c;
    const __bf16* gBrow = Bp + (size_t)(n0 + r) * ldb + c;

    // issue one k-tile (per wave: 4 async b128 instructions -> ASYNCcnt += 4)
    auto issue = [&](int buf, int k0) {
        ASYNC_B128(gArow + k0,     &As[buf][r][c]);
        ASYNC_B128(gArow + k0 + 8, &As[buf][r][c + 8]);
        ASYNC_B128(gBrow + k0,     &Bs[buf][r][c]);
        ASYNC_B128(gBrow + k0 + 8, &Bs[buf][r][c + 8]);
    };

    // fragments per documented gfx1250 16-bit WMMA layouts + 8 WMMAs
    auto compute = [&](int buf) {
        // A 16x32: lanes0-15 M=lr K={0..7,16..23}; lanes16-31 K={8..15,24..31}
        Frag af[4];
        const int akb = half * 8;
        #pragma unroll
        for (int i = 0; i < 4; ++i) {
            const int row = wrow * 64 + i * 16 + lr;
            af[i].h[0] = *(const v8bf*)&As[buf][row][akb];
            af[i].h[1] = *(const v8bf*)&As[buf][row][akb + 16];
        }
        // B 32x16: lanes0-15 N=lr K={0..7,16..23}; lanes16-31 K={8..15,24..31}
        Frag bfr[2];
        const int bkb = half * 8;
        #pragma unroll
        for (int j = 0; j < 2; ++j) {
            const int col = wcol * 32 + j * 16 + lr;
            bfr[j].h[0] = *(const v8bf*)&Bs[buf][col][bkb];
            bfr[j].h[1] = *(const v8bf*)&Bs[buf][col][bkb + 16];
        }
        #pragma unroll
        for (int i = 0; i < 4; ++i)
            #pragma unroll
            for (int j = 0; j < 2; ++j)
                acc[i][j] = __builtin_amdgcn_wmma_f32_16x16x32_bf16(
                    false, af[i].v, false, bfr[j].v,
                    (short)0, acc[i][j], false, false);
    };

    const int nsteps = K / BK;           // 16 or 25, always >= 2
    issue(0, 0);
    issue(1, BK);

    int s = 0;
    for (; s < nsteps - 2; ++s) {        // branch-free steady state
        WAIT_ASYNC(4);                   // this buffer done; next 4 in flight
        __syncthreads();
        compute(s & 1);
        __syncthreads();
        issue(s & 1, (s + 2) * BK);
    }
    WAIT_ASYNC(4);                       // penultimate step: no more issues
    __syncthreads();
    compute(s & 1);
    __syncthreads();
    ++s;
    WAIT_ASYNC(0);                       // last step: drain
    __syncthreads();
    compute(s & 1);

    // ---- epilogue: alpha/beta, bf16 store, optional Frobenius normsq ----
    // C/D layout: VGPR g -> M = g + 8*half ; N = lr
    float local = 0.f;
    #pragma unroll
    for (int i = 0; i < 4; ++i)
        #pragma unroll
        for (int j = 0; j < 2; ++j)
            #pragma unroll
            for (int g = 0; g < 8; ++g) {
                const int m = m0 + wrow * 64 + i * 16 + half * 8 + g;
                const int n = n0 + wcol * 32 + j * 16 + lr;
                float val = alpha * acc[i][j][g];
                if (m == n) val += beta;
                C[(size_t)m * DIM + n] = (__bf16)val;
                local += val * val;
            }
    if (normsq) {
        red[tid] = local;
        __syncthreads();
        for (int st = 128; st > 0; st >>= 1) {
            if (tid < st) red[tid] += red[tid + st];
            __syncthreads();
        }
        if (tid == 0) atomicAdd(&normsq[b], red[0]);
    }
}

// ---- X (f32, K=784) -> Xbf (bf16, K padded to 800 with zeros) --------------
__global__ void convert_pad_kernel(const float* __restrict__ X,
                                   __bf16* __restrict__ Xb, long total)
{
    long i = (long)blockIdx.x * blockDim.x + threadIdx.x;
    if (i >= total) return;
    const int  p  = (int)(i % KPAD);
    const long rd = i / KPAD;
    const float v = (p < SPATIAL) ? X[rd * SPATIAL + p] : 0.f;
    Xb[i] = (__bf16)v;
}

// ---- zero reductions, out = bias ------------------------------------------
__global__ void init_misc_kernel(float* __restrict__ normsq, float* __restrict__ vsumsq,
                                 float* __restrict__ out, const float* __restrict__ bias)
{
    const int i = blockIdx.x * blockDim.x + threadIdx.x;
    if (i < NBATCH) { normsq[i] = 0.f; vsumsq[i] = 0.f; }
    if (i < NBATCH * NOUT) out[i] = bias[i % NOUT];
}

// ---- Y = A/||A||_F (in place, bf16) ; Z = I --------------------------------
__global__ void ns_init_kernel(__bf16* __restrict__ Y, __bf16* __restrict__ Z,
                               const float* __restrict__ normsq)
{
    const long total = (long)NBATCH * DIM * DIM;
    long i = (long)blockIdx.x * blockDim.x + threadIdx.x;
    if (i >= total) return;
    const int b  = (int)(i / ((long)DIM * DIM));
    const int rc = (int)(i % ((long)DIM * DIM));
    const float inv = 1.f / sqrtf(fmaxf(normsq[b], 1e-30f));
    Y[i] = (__bf16)((float)Y[i] * inv);
    Z[i] = (__bf16)(((rc / DIM) == (rc % DIM)) ? 1.f : 0.f);
}

// ---- per-batch sum(v^2), v = sign(w)*sqrt(|w|+1e-5), w = Y*||A||^0.5 -------
__global__ void v_sumsq_kernel(const __bf16* __restrict__ Y,
                               const float* __restrict__ normsq,
                               float* __restrict__ vsumsq)
{
    __shared__ float red[256];
    const int  b     = blockIdx.y;
    const long base  = (long)b * DIM * DIM + (long)blockIdx.x * 4096;
    const float scale = sqrtf(sqrtf(fmaxf(normsq[b], 1e-30f)));  // normsq^(1/4)
    float s = 0.f;
    for (int t = threadIdx.x; t < 4096; t += 256) {
        const float w = (float)Y[base + t] * scale;
        const float v = copysignf(sqrtf(fabsf(w) + 1e-5f), w);
        s += v * v;
    }
    red[threadIdx.x] = s; __syncthreads();
    for (int st = 128; st > 0; st >>= 1) {
        if (threadIdx.x < st) red[threadIdx.x] += red[threadIdx.x + st];
        __syncthreads();
    }
    if (threadIdx.x == 0) atomicAdd(&vsumsq[b], red[0]);
}

// ---- out[b,o] += (1/max(||v||,1e-12)) * sum_k v[b,k]*W[o,k]  (split-K) -----
#define FCCHUNK 8192
__global__ __launch_bounds__(256)
void fc_kernel(const __bf16* __restrict__ Y, const float* __restrict__ W,
               const float* __restrict__ normsq, const float* __restrict__ vsumsq,
               float* __restrict__ out)
{
    __shared__ float Wl[FCCHUNK];       // 32 KB of one fc_weight row chunk
    __shared__ float red[256];
    const int  o  = blockIdx.x;
    const long kb = (long)blockIdx.y * FCCHUNK;
    for (int t = threadIdx.x; t < FCCHUNK; t += 256)
        Wl[t] = W[(long)o * ((long)DIM * DIM) + kb + t];
    __syncthreads();
    for (int b = 0; b < NBATCH; ++b) {
        const float scale = sqrtf(sqrtf(fmaxf(normsq[b], 1e-30f)));
        const __bf16* yb = Y + (long)b * DIM * DIM + kb;
        float s = 0.f;
        for (int t = threadIdx.x; t < FCCHUNK; t += 256) {
            const float w = (float)yb[t] * scale;
            const float v = copysignf(sqrtf(fabsf(w) + 1e-5f), w);
            s += v * Wl[t];
        }
        red[threadIdx.x] = s; __syncthreads();
        for (int st = 128; st > 0; st >>= 1) {
            if (threadIdx.x < st) red[threadIdx.x] += red[threadIdx.x + st];
            __syncthreads();
        }
        if (threadIdx.x == 0) {
            const float n = fmaxf(sqrtf(vsumsq[b]), 1e-12f);
            atomicAdd(&out[b * NOUT + o], red[0] / n);
        }
        __syncthreads();
    }
}

// ---------------------------------------------------------------------------
extern "C" void kernel_launch(void* const* d_in, const int* in_sizes, int n_in,
                              void* d_out, int out_size, void* d_ws, size_t ws_size,
                              hipStream_t stream)
{
    (void)in_sizes; (void)n_in; (void)out_size; (void)ws_size;
    const float* X    = (const float*)d_in[0];   // (64, 512, 28, 28) f32
    const float* W    = (const float*)d_in[1];   // (200, 262144) f32
    const float* bias = (const float*)d_in[2];   // (200,) f32
    float* out = (float*)d_out;                  // (64, 200) f32

    // workspace carve-out (~220 MB)
    char* p = (char*)d_ws;
    auto take = [&](size_t bytes) -> char* {
        char* r = p; p += (bytes + 255) & ~(size_t)255; return r;
    };
    const size_t MATB = (size_t)NBATCH * DIM * DIM * sizeof(__bf16);   // 33.5 MB
    __bf16* Xb      = (__bf16*)take((size_t)NBATCH * DIM * KPAD * sizeof(__bf16));
    __bf16* Ybuf[2] = { (__bf16*)take(MATB), (__bf16*)take(MATB) };
    __bf16* Zbuf[2] = { (__bf16*)take(MATB), (__bf16*)take(MATB) };
    __bf16* Tb      = (__bf16*)take(MATB);
    float*  normsq  = (float*)take(256);
    float*  vsumsq  = (float*)take(256);

    init_misc_kernel<<<(NBATCH * NOUT + 255) / 256, 256, 0, stream>>>(normsq, vsumsq, out, bias);

    const long ctot = (long)NBATCH * DIM * KPAD;
    convert_pad_kernel<<<(unsigned)((ctot + 255) / 256), 256, 0, stream>>>(X, Xb, ctot);

    // Gram: A = (1/784) * x x^T  -> Ybuf[0] (bf16) + Frobenius normsq
    gemm_bf16_wmma<<<dim3(DIM / BN, DIM / BM, NBATCH), 256, 0, stream>>>(
        Xb, nullptr, Xb, Ybuf[0], nullptr,
        NBATCH, KPAD, KPAD, KPAD, 1.f / (float)SPATIAL, 0.f, normsq);

    const long mtot = (long)NBATCH * DIM * DIM;
    ns_init_kernel<<<(unsigned)((mtot + 255) / 256), 256, 0, stream>>>(Ybuf[0], Zbuf[0], normsq);

    // Newton-Schulz: T = 1.5I - 0.5 Z*Y ; Ynew = Y*T ; Znew = T*Z (= Z*T)
    int cur = 0;
    for (int it = 0; it < 10; ++it) {
        gemm_bf16_wmma<<<dim3(DIM / BN, DIM / BM, NBATCH), 256, 0, stream>>>(
            Zbuf[cur], nullptr, Ybuf[cur], Tb, nullptr,
            NBATCH, DIM, DIM, DIM, -0.5f, 1.5f, nullptr);
        gemm_bf16_wmma<<<dim3(DIM / BN, DIM / BM, 2 * NBATCH), 256, 0, stream>>>(
            Ybuf[cur], Zbuf[cur], Tb, Ybuf[1 - cur], Zbuf[1 - cur],
            NBATCH, DIM, DIM, DIM, 1.f, 0.f, nullptr);
        cur ^= 1;
    }

    v_sumsq_kernel<<<dim3((DIM * DIM) / 4096, NBATCH), 256, 0, stream>>>(Ybuf[cur], normsq, vsumsq);
    fc_kernel<<<dim3(NOUT, (DIM * DIM) / FCCHUNK), 256, 0, stream>>>(Ybuf[cur], W, normsq, vsumsq, out);
}
